// HeadOptimal_78219944394812
// MI455X (gfx1250) — compile-verified, matrix operand order checked by reading, not run on hardware
//
#include <hip/hip_runtime.h>
#include <cstddef>
#include <cstdint>

// Problem constants (match reference)
#define B_N 16
#define P_N 512
#define S_N 1024
#define D_N 1024
#define H_FULL 16
#define H_LIGHT 8

typedef __attribute__((ext_vector_type(16))) _Float16 v16h;
typedef __attribute__((ext_vector_type(8)))  float    v8f;
typedef __attribute__((ext_vector_type(4)))  int      v4i;

// ---------------------------------------------------------------------------
// WMMA fragment helpers (16x16x32 f16, wave32).
// 16-bit A/B operand layout (ISA 7.12.2): lanes 0-15 hold outer index 0-15,
// VGPR j holds K pair (j<4 ? 2j : 16+2(j-4)) + (lane>=16 ? 8 : 0).
// ---------------------------------------------------------------------------
__device__ __forceinline__ int frag_koff(int p, int hi) {
  return (p < 4 ? 2 * p : 16 + 2 * (p - 4)) + (hi ? 8 : 0);
}

__device__ __forceinline__ v16h load_frag_f32row(const float* row, int kbase, int hi) {
  v16h f;
#pragma unroll
  for (int p = 0; p < 8; ++p) {
    int k = kbase + frag_koff(p, hi);
    float2 t = *(const float2*)(row + k);
    f[2 * p]     = (_Float16)t.x;
    f[2 * p + 1] = (_Float16)t.y;
  }
  return f;
}

__device__ __forceinline__ v8f zero_v8f() {
  v8f z;
#pragma unroll
  for (int i = 0; i < 8; ++i) z[i] = 0.0f;
  return z;
}

// ---------------------------------------------------------------------------
// CDNA5 async global->LDS copy (GLOBAL_LOAD_ASYNC_TO_LDS_B128, ASYNCcnt).
// Builtin expects v4i pointers: global src in AS1, LDS dst in AS3.
// Generic LDS address low 32 bits == LDS byte offset (ISA 10.2 aperture calc).
// ---------------------------------------------------------------------------
typedef __attribute__((address_space(1))) v4i* gbl_v4i_p;
typedef __attribute__((address_space(3))) v4i* lds_v4i_p;

__device__ __forceinline__ void async_copy_b128(const float* g, float* l) {
#if __has_builtin(__builtin_amdgcn_global_load_async_to_lds_b128)
  __builtin_amdgcn_global_load_async_to_lds_b128(
      (gbl_v4i_p)(uintptr_t)g, (lds_v4i_p)(uint32_t)(uintptr_t)l, 0, 0);
#else
  uint32_t lds_off = (uint32_t)(uintptr_t)l;
  asm volatile("global_load_async_to_lds_b128 %0, %1, off"
               :: "v"(lds_off), "v"(g) : "memory");
#endif
}

__device__ __forceinline__ void wait_async_all() {
#if __has_builtin(__builtin_amdgcn_s_wait_asynccnt)
  __builtin_amdgcn_s_wait_asynccnt(0);
#else
  asm volatile("s_wait_asynccnt 0x0" ::: "memory");
#endif
}

// ---------------------------------------------------------------------------
// GEMM: C[M,N] = A[M,K] @ W[N,K]^T + bias[N]  (+ optional exact GELU)
// Workgroup = 256 threads (8 waves) computing a 128x128 tile; A/W K-panels
// async-copied into LDS, each wave does a 16(M) x 128(N) slice (8 WMMAs/step).
// M % 128 == 0, N % 128 == 0, K % 32 == 0.
// ---------------------------------------------------------------------------
#define GBM 128
#define GBN 128
#define GBK 32

__global__ __launch_bounds__(256) void wmma_gemm_bias(
    const float* __restrict__ A, const float* __restrict__ W,
    const float* __restrict__ bias, float* __restrict__ C,
    int M, int N, int K, int act) {
  __shared__ float As[GBM * GBK];  // 16 KB
  __shared__ float Ws[GBN * GBK];  // 16 KB

  const int tid  = threadIdx.x;
  const int lane = tid & 31;
  const int wv   = tid >> 5;       // wave id 0..7 -> M sub-tile
  const int hi   = (lane >> 4) & 1;
  const int lrow = lane & 15;
  const int n0   = blockIdx.x * GBN;
  const int m0   = blockIdx.y * GBM;
  (void)M;

  v8f acc[8];
#pragma unroll
  for (int c = 0; c < 8; ++c) acc[c] = zero_v8f();

  for (int kb = 0; kb < K; kb += GBK) {
    // Stage A panel: 128x32 f32 = 1024 x 16B chunks (4 per thread).
#pragma unroll
    for (int i = 0; i < (GBM * GBK) / (4 * 256); ++i) {
      int q = tid + i * 256;                 // chunk id
      int r = q >> 3, cc = (q & 7) * 4;      // row, col (floats)
      async_copy_b128(A + (size_t)(m0 + r) * K + kb + cc, &As[r * GBK + cc]);
    }
    // Stage W panel: 128x32 f32.
#pragma unroll
    for (int i = 0; i < (GBN * GBK) / (4 * 256); ++i) {
      int q = tid + i * 256;
      int r = q >> 3, cc = (q & 7) * 4;
      async_copy_b128(W + (size_t)(n0 + r) * K + kb + cc, &Ws[r * GBK + cc]);
    }
    wait_async_all();
    __syncthreads();

    v16h a = load_frag_f32row(&As[(wv * 16 + lrow) * GBK], 0, hi);
#pragma unroll
    for (int c = 0; c < 8; ++c) {
      v16h b = load_frag_f32row(&Ws[(c * 16 + lrow) * GBK], 0, hi);
      acc[c] = __builtin_amdgcn_wmma_f32_16x16x32_f16(false, a, false, b,
                                                      (short)0, acc[c], false, false);
    }
    __syncthreads();
  }

#pragma unroll
  for (int c = 0; c < 8; ++c) {
    int n = n0 + c * 16 + lrow;
    float bn = bias[n];
#pragma unroll
    for (int r = 0; r < 8; ++r) {
      int m = m0 + wv * 16 + r + (hi ? 8 : 0);
      float v = acc[c][r] + bn;
      if (act == 1) v = 0.5f * v * (1.0f + erff(v * 0.70710678118654752f));
      C[(size_t)m * N + n] = v;
    }
  }
}

// ---------------------------------------------------------------------------
// Attention: one wave per (batch, head, 16-query tile). Full softmax row kept
// in LDS. Optionally accumulates mean-over-heads attention weights (atomic).
// HD in {64,128}; LK = key length (compile-time so static LDS sizes).
// ---------------------------------------------------------------------------
template <int HD, int LK>
__global__ void attention_kernel(const float* __restrict__ Q, const float* __restrict__ Kt,
                                 const float* __restrict__ V, float* __restrict__ O,
                                 float* __restrict__ attnw, int Lq, int H,
                                 float scale, float aw_scale) {
  __shared__ float sm[16 * LK];   // probabilities / scores, fp32

  const int lane = threadIdx.x;
  const int hi   = lane >> 4;
  const int lrow = lane & 15;
  const int q0   = blockIdx.x * 16;
  const int h    = blockIdx.y;
  const int b    = blockIdx.z;
  const int Dm   = H * HD;
  constexpr int NK   = HD / 32;  // K-chunks per dot product
  constexpr int NCOL = HD / 16;  // 16-wide output column groups

  // Preload Q fragments for this tile (row lrow of the tile, head h).
  const float* qrow = Q + ((size_t)b * Lq + q0 + lrow) * Dm + h * HD;
  v16h qf[NK];
#pragma unroll
  for (int kc = 0; kc < NK; ++kc) qf[kc] = load_frag_f32row(qrow, kc * 32, hi);

  // Scores -> LDS
  for (int t = 0; t < LK; t += 16) {
    const float* krow = Kt + ((size_t)b * LK + t + lrow) * Dm + h * HD;
    v8f acc = zero_v8f();
#pragma unroll
    for (int kc = 0; kc < NK; ++kc) {
      v16h kf = load_frag_f32row(krow, kc * 32, hi);
      acc = __builtin_amdgcn_wmma_f32_16x16x32_f16(false, qf[kc], false, kf,
                                                   (short)0, acc, false, false);
    }
#pragma unroll
    for (int r = 0; r < 8; ++r) {
      int m = r + (hi ? 8 : 0);
      sm[m * LK + t + lrow] = acc[r] * scale;
    }
  }
  __syncthreads();

  // Softmax: lanes 0-15 each own one full row.
  if (lane < 16) {
    float* row = sm + lrow * LK;
    float mx = -3.402823466e38f;
    for (int c = 0; c < LK; ++c) mx = fmaxf(mx, row[c]);
    float sum = 0.0f;
    for (int c = 0; c < LK; ++c) { float e = expf(row[c] - mx); row[c] = e; sum += e; }
    float inv = 1.0f / sum;
    for (int c = 0; c < LK; ++c) row[c] *= inv;
  }
  __syncthreads();

  // Mean-over-heads attention weights (cross attention only).
  if (attnw != nullptr) {
    float* dst = attnw + ((size_t)b * Lq + q0 + lrow) * LK;
    const float* row = sm + lrow * LK;
    for (int c = hi * (LK / 2); c < (hi + 1) * (LK / 2); ++c)
      atomicAdd(&dst[c], row[c] * aw_scale);
  }

  // O = P @ V
  v8f oacc[NCOL];
#pragma unroll
  for (int c = 0; c < NCOL; ++c) oacc[c] = zero_v8f();

  for (int kb = 0; kb < LK; kb += 32) {
    v16h pf = load_frag_f32row(sm + lrow * LK, kb, hi);
#pragma unroll
    for (int c = 0; c < NCOL; ++c) {
      const float* vcol = V + ((size_t)b * LK + kb) * Dm + h * HD + c * 16 + lrow;
      v16h vf;
#pragma unroll
      for (int p = 0; p < 8; ++p) {
        int k = frag_koff(p, hi);
        vf[2 * p]     = (_Float16)vcol[(size_t)k * Dm];
        vf[2 * p + 1] = (_Float16)vcol[(size_t)(k + 1) * Dm];
      }
      oacc[c] = __builtin_amdgcn_wmma_f32_16x16x32_f16(false, pf, false, vf,
                                                       (short)0, oacc[c], false, false);
    }
  }

#pragma unroll
  for (int c = 0; c < NCOL; ++c) {
#pragma unroll
    for (int r = 0; r < 8; ++r) {
      int m = r + (hi ? 8 : 0);
      O[((size_t)b * Lq + q0 + m) * Dm + h * HD + c * 16 + lrow] = oacc[c][r];
    }
  }
}

// ---------------------------------------------------------------------------
// y = LayerNorm(x + res) * g + b ; one 256-thread block per row, Dm == 1024.
// ---------------------------------------------------------------------------
__global__ void ln_residual_kernel(const float* __restrict__ x, const float* __restrict__ res,
                                   const float* __restrict__ g, const float* __restrict__ bb,
                                   float* __restrict__ y, int Dm) {
  const int row = blockIdx.x;
  const int tid = threadIdx.x;
  const int per = Dm / 256;  // 4
  __shared__ float s1[256], s2[256];

  const float* xr = x + (size_t)row * Dm;
  const float* rr = res + (size_t)row * Dm;
  float v[4];
  float sum = 0.0f, sq = 0.0f;
#pragma unroll
  for (int i = 0; i < 4; ++i) {
    if (i < per) {
      int d = tid + i * 256;
      float t = xr[d] + rr[d];
      v[i] = t; sum += t; sq += t * t;
    }
  }
  s1[tid] = sum; s2[tid] = sq;
  __syncthreads();
  for (int off = 128; off > 0; off >>= 1) {
    if (tid < off) { s1[tid] += s1[tid + off]; s2[tid] += s2[tid + off]; }
    __syncthreads();
  }
  float mu   = s1[0] / (float)Dm;
  float var  = s2[0] / (float)Dm - mu * mu;
  float rstd = rsqrtf(var + 1e-5f);
#pragma unroll
  for (int i = 0; i < 4; ++i) {
    if (i < per) {
      int d = tid + i * 256;
      y[(size_t)row * Dm + d] = (v[i] - mu) * rstd * g[d] + bb[d];
    }
  }
}

// gate_in = [a | b | c]  (each [M,Dm] -> [M,3*Dm])
__global__ void concat3_kernel(const float* __restrict__ a, const float* __restrict__ b,
                               const float* __restrict__ c, float* __restrict__ out,
                               size_t total, int Dm) {
  size_t i = (size_t)blockIdx.x * blockDim.x + threadIdx.x;
  if (i >= total) return;
  size_t m = i / Dm, d = i % Dm;
  out[m * (3 * Dm) + d]          = a[i];
  out[m * (3 * Dm) + Dm + d]     = b[i];
  out[m * (3 * Dm) + 2 * Dm + d] = c[i];
}

// logits = h @ w2^T + b2 (3 outputs), softmax, gated combine of c0/c1/c2.
__global__ void gate_final_kernel(const float* __restrict__ h, const float* __restrict__ w2,
                                  const float* __restrict__ b2,
                                  const float* __restrict__ c0, const float* __restrict__ c1,
                                  const float* __restrict__ c2, float* __restrict__ out, int Dm) {
  const int row = blockIdx.x;
  const int tid = threadIdx.x;  // 128 threads
  __shared__ float r0[128], r1[128], r2[128];
  __shared__ float gw[3];

  const float* hr = h + (size_t)row * Dm;
  float p0 = 0.f, p1 = 0.f, p2 = 0.f;
  for (int d = tid; d < Dm; d += 128) {
    float hv = hr[d];
    p0 += hv * w2[d];
    p1 += hv * w2[Dm + d];
    p2 += hv * w2[2 * Dm + d];
  }
  r0[tid] = p0; r1[tid] = p1; r2[tid] = p2;
  __syncthreads();
  for (int off = 64; off > 0; off >>= 1) {
    if (tid < off) { r0[tid] += r0[tid + off]; r1[tid] += r1[tid + off]; r2[tid] += r2[tid + off]; }
    __syncthreads();
  }
  if (tid == 0) {
    float l0 = r0[0] + b2[0], l1 = r1[0] + b2[1], l2 = r2[0] + b2[2];
    float mx = fmaxf(l0, fmaxf(l1, l2));
    float e0 = expf(l0 - mx), e1 = expf(l1 - mx), e2 = expf(l2 - mx);
    float inv = 1.0f / (e0 + e1 + e2);
    gw[0] = e0 * inv; gw[1] = e1 * inv; gw[2] = e2 * inv;
  }
  __syncthreads();
  float g0 = gw[0], g1 = gw[1], g2 = gw[2];
  for (int d = tid; d < Dm; d += 128) {
    size_t i = (size_t)row * Dm + d;
    out[i] = g0 * c0[i] + g1 * c1[i] + g2 * c2[i];
  }
}

__global__ void zero_kernel(float* __restrict__ p, size_t n) {
  size_t i = (size_t)blockIdx.x * blockDim.x + threadIdx.x;
  if (i < n) p[i] = 0.0f;
}

// ---------------------------------------------------------------------------
static inline void launch_gemm(const float* A, const float* W, const float* bias, float* C,
                               int M, int N, int K, int act, hipStream_t s) {
  dim3 grid(N / GBN, M / GBM), blk(256);
  wmma_gemm_bias<<<grid, blk, 0, s>>>(A, W, bias, C, M, N, K, act);
}

extern "C" void kernel_launch(void* const* d_in, const int* in_sizes, int n_in,
                              void* d_out, int out_size, void* d_ws, size_t ws_size,
                              hipStream_t stream) {
  (void)in_sizes; (void)n_in; (void)out_size; (void)ws_size;

  const float* proto = (const float*)d_in[0];
  const float* img   = (const float*)d_in[1];
  const float* cw_in = (const float*)d_in[2];
  const float* cb_in = (const float*)d_in[3];
  const float* cw_o  = (const float*)d_in[4];
  const float* cb_o  = (const float*)d_in[5];
  const float* sw_in = (const float*)d_in[6];
  const float* sb_in = (const float*)d_in[7];
  const float* sw_o  = (const float*)d_in[8];
  const float* sb_o  = (const float*)d_in[9];
  const float* lw_in = (const float*)d_in[10];
  const float* lb_in = (const float*)d_in[11];
  const float* lw_o  = (const float*)d_in[12];
  const float* lb_o  = (const float*)d_in[13];
  const float* n1g = (const float*)d_in[14];
  const float* n1b = (const float*)d_in[15];
  const float* n2g = (const float*)d_in[16];
  const float* n2b = (const float*)d_in[17];
  const float* n3g = (const float*)d_in[18];
  const float* n3b = (const float*)d_in[19];
  const float* gw1 = (const float*)d_in[20];
  const float* gb1 = (const float*)d_in[21];
  const float* gw2 = (const float*)d_in[22];
  const float* gb2 = (const float*)d_in[23];

  const int BP = B_N * P_N;              // 8192
  const int BS = B_N * S_N;              // 16384
  const size_t M8  = (size_t)BP * D_N;   // 8,388,608
  const size_t M16 = (size_t)BS * D_N;   // 16,777,216
  const size_t DD  = (size_t)D_N * D_N;

  float* out_upd  = (float*)d_out;
  float* out_attn = (float*)d_out + (size_t)B_N * P_N * S_N;  // second return value

  // Workspace layout (fp32), total 112M floats = 448 MB.
  float* ws     = (float*)d_ws;
  float* Qb     = ws;
  float* Kb     = Qb + M8;
  float* Vb     = Kb + M16;
  float* attO   = Vb + M16;
  float* projB  = attO + M8;
  float* crossO = projB + M8;
  float* selfO  = crossO + M8;
  float* lightO = selfO + M8;
  float* gateIn = lightO + M8;
  float* hbuf   = gateIn + 3 * M8;

  // attn_w is accumulated atomically -> zero it each call (deterministic).
  {
    size_t n = (size_t)B_N * P_N * S_N;
    zero_kernel<<<(unsigned)((n + 255) / 256), 256, 0, stream>>>(out_attn, n);
  }

  // ---- Stage 1: cross attention (16 heads, hd=64, keys = img S=1024) ----
  launch_gemm(proto, cw_in,          cb_in,           Qb, BP, D_N, D_N, 0, stream);
  launch_gemm(img,   cw_in + DD,     cb_in + D_N,     Kb, BS, D_N, D_N, 0, stream);
  launch_gemm(img,   cw_in + 2 * DD, cb_in + 2 * D_N, Vb, BS, D_N, D_N, 0, stream);
  attention_kernel<64, 1024><<<dim3(P_N / 16, H_FULL, B_N), 32, 0, stream>>>(
      Qb, Kb, Vb, attO, out_attn, P_N, H_FULL, 0.125f, 1.0f / (float)H_FULL);
  launch_gemm(attO, cw_o, cb_o, projB, BP, D_N, D_N, 0, stream);
  ln_residual_kernel<<<BP, 256, 0, stream>>>(projB, proto, n1g, n1b, crossO, D_N);

  // ---- Stage 2: self attention (16 heads, hd=64, keys = P=512) ----
  launch_gemm(crossO, sw_in,          sb_in,           Qb, BP, D_N, D_N, 0, stream);
  launch_gemm(crossO, sw_in + DD,     sb_in + D_N,     Kb, BP, D_N, D_N, 0, stream);
  launch_gemm(crossO, sw_in + 2 * DD, sb_in + 2 * D_N, Vb, BP, D_N, D_N, 0, stream);
  attention_kernel<64, 512><<<dim3(P_N / 16, H_FULL, B_N), 32, 0, stream>>>(
      Qb, Kb, Vb, attO, nullptr, P_N, H_FULL, 0.125f, 0.0f);
  launch_gemm(attO, sw_o, sb_o, projB, BP, D_N, D_N, 0, stream);
  ln_residual_kernel<<<BP, 256, 0, stream>>>(projB, crossO, n2g, n2b, selfO, D_N);

  // ---- Stage 3: light attention (8 heads, hd=128, keys = P=512) ----
  launch_gemm(selfO, lw_in,          lb_in,           Qb, BP, D_N, D_N, 0, stream);
  launch_gemm(selfO, lw_in + DD,     lb_in + D_N,     Kb, BP, D_N, D_N, 0, stream);
  launch_gemm(selfO, lw_in + 2 * DD, lb_in + 2 * D_N, Vb, BP, D_N, D_N, 0, stream);
  attention_kernel<128, 512><<<dim3(P_N / 16, H_LIGHT, B_N), 32, 0, stream>>>(
      Qb, Kb, Vb, attO, nullptr, P_N, H_LIGHT, 0.08838834764831845f, 0.0f);
  launch_gemm(attO, lw_o, lb_o, projB, BP, D_N, D_N, 0, stream);
  ln_residual_kernel<<<BP, 256, 0, stream>>>(projB, selfO, n3g, n3b, lightO, D_N);

  // ---- Stage 4: gated fusion ----
  concat3_kernel<<<(unsigned)((M8 + 255) / 256), 256, 0, stream>>>(
      crossO, selfO, lightO, gateIn, M8, D_N);
  launch_gemm(gateIn, gw1, gb1, hbuf, BP, D_N, 3 * D_N, /*gelu*/ 1, stream);
  gate_final_kernel<<<BP, 128, 0, stream>>>(hbuf, gw2, gb2, crossO, selfO, lightO,
                                            out_upd, D_N);
}
